// VitPoseBackboneMoeMLP_33346126086454
// MI455X (gfx1250) — compile-verified, matrix operand order checked by reading, not run on hardware
//
#include <hip/hip_runtime.h>
#include <hip/hip_bf16.h>
#include <math.h>

// ---- problem constants (from reference) ----
#define BDIM   64
#define SDIM   192
#define DDIM   768     // model dim (= GEMM1 K, = output N)
#define HDIM   3072    // hidden (= GEMM1 N, = GEMM2 K)
#define PDIM   256     // expert out features
#define SHRD   512     // shared fc2 out features
#define MTILE  32      // rows of (b,s) per workgroup
#define HC     128     // H-chunk width staged in LDS
#define XS     (DDIM + 8)   // LDS stride for x tile (bf16 elems)
#define HS     (HC + 8)     // LDS stride for h chunk (bf16 elems)

typedef __attribute__((ext_vector_type(8)))  float  v8f;
typedef __attribute__((ext_vector_type(16))) __bf16 v16bf;
typedef __attribute__((ext_vector_type(8)))  __bf16 v8bf;

// fp32 -> bf16 round-to-nearest-even (used only in staging / fallback paths)
__device__ __forceinline__ __bf16 f2bf(float f) {
    unsigned u = __builtin_bit_cast(unsigned, f);
    u += 0x7FFFu + ((u >> 16) & 1u);
    unsigned short s = (unsigned short)(u >> 16);
    return __builtin_bit_cast(__bf16, s);
}

// Fragment layout (ISA 16-bit A/B): lane half 0 carries K {k0..k0+7, k0+16..k0+23},
// half 1 carries K {k0+8..k0+15, k0+24..k0+31} -> two contiguous 16B chunks when the
// source is bf16 row-major along K. p already includes row*stride + half*8.
__device__ __forceinline__ v16bf load_wfrag(const __bf16* p) {
    v8bf lo = *(const v8bf*)(p);
    v8bf hi = *(const v8bf*)(p + 16);
    return __builtin_shufflevector(lo, hi, 0, 1, 2, 3, 4, 5, 6, 7,
                                   8, 9, 10, 11, 12, 13, 14, 15);
}

// fallback: fragment from fp32 weights, converting inline
__device__ __forceinline__ v16bf load_wfrag(const float* p) {
    float4 a0 = *(const float4*)(p);
    float4 a1 = *(const float4*)(p + 4);
    float4 b0 = *(const float4*)(p + 16);
    float4 b1 = *(const float4*)(p + 20);
    v16bf r;
    r[0]  = f2bf(a0.x); r[1]  = f2bf(a0.y); r[2]  = f2bf(a0.z); r[3]  = f2bf(a0.w);
    r[4]  = f2bf(a1.x); r[5]  = f2bf(a1.y); r[6]  = f2bf(a1.z); r[7]  = f2bf(a1.w);
    r[8]  = f2bf(b0.x); r[9]  = f2bf(b0.y); r[10] = f2bf(b0.z); r[11] = f2bf(b0.w);
    r[12] = f2bf(b1.x); r[13] = f2bf(b1.y); r[14] = f2bf(b1.z); r[15] = f2bf(b1.w);
    return r;
}

// Branch-free exact-erf GELU (Abramowitz–Stegun 7.1.26, |err| < 1.5e-7):
// one v_rcp_f32 + one v_exp_f32, no EXEC divergence.
__device__ __forceinline__ float gelu_erf(float v) {
    float z  = v * 0.70710678118654752f;
    float az = __builtin_fabsf(z);
    float t  = __builtin_amdgcn_rcpf(__builtin_fmaf(0.3275911f, az, 1.0f));
    float p  = __builtin_fmaf(t, 1.061405429f, -1.453152027f);
    p = __builtin_fmaf(t, p, 1.421413741f);
    p = __builtin_fmaf(t, p, -0.284496736f);
    p = __builtin_fmaf(t, p, 0.254829592f);
    p = p * t;
    float e  = __builtin_amdgcn_exp2f(-az * az * 1.4426950408889634f); // e^{-z^2}
    float ea = __builtin_fmaf(-p, e, 1.0f);                            // erf(|z|)
    float er = __builtin_copysignf(ea, z);
    return 0.5f * v * (1.0f + er);
}

// elementwise fp32 -> bf16 (8 elems/thread); n8 = n/8
__global__ __launch_bounds__(256)
void cvt_f32_to_bf16(const float* __restrict__ src, __bf16* __restrict__ dst, int n8) {
    int i = blockIdx.x * 256 + threadIdx.x;
    if (i >= n8) return;
    float4 a = ((const float4*)src)[2 * i];
    float4 b = ((const float4*)src)[2 * i + 1];
    v8bf o;
    o[0] = f2bf(a.x); o[1] = f2bf(a.y); o[2] = f2bf(a.z); o[3] = f2bf(a.w);
    o[4] = f2bf(b.x); o[5] = f2bf(b.y); o[6] = f2bf(b.z); o[7] = f2bf(b.w);
    *(v8bf*)(dst + (size_t)i * 8) = o;
}

template <typename WT>
__global__ __launch_bounds__(256, 1)
void moe_mlp_fused(const float* __restrict__ x,
                   const long long* __restrict__ indices,
                   const WT* __restrict__ w1, const float* __restrict__ b1,
                   const WT* __restrict__ w2, const float* __restrict__ b2,
                   const WT* __restrict__ we, const float* __restrict__ be,
                   float* __restrict__ out) {
    __shared__ __bf16 xs[MTILE * XS];   // staged input tile, bf16
    __shared__ __bf16 hs[MTILE * HS];   // current GELU(h) chunk, bf16

    const int tid  = threadIdx.x;
    const int wave = tid >> 5;
    const int lane = tid & 31;
    const int ln   = lane & 15;   // row/col within 16
    const int half = lane >> 4;   // which K-half this lane carries

    const int wg    = blockIdx.x;
    const int b     = wg / (SDIM / MTILE);
    const int stile = wg % (SDIM / MTILE);
    const int e     = (int)indices[b];

    // ---- stage x tile fp32 -> bf16 in LDS (32 x 768) ----
    const float* xsrc = x + (size_t)(b * SDIM + stile * MTILE) * DDIM;
    for (int i = tid; i < (MTILE * DDIM) / 4; i += 256) {
        float4 v = ((const float4*)xsrc)[i];
        int r = i / (DDIM / 4);
        int c = (i % (DDIM / 4)) * 4;
        __bf16* d = &xs[r * XS + c];
        d[0] = f2bf(v.x); d[1] = f2bf(v.y); d[2] = f2bf(v.z); d[3] = f2bf(v.w);
    }

    // ---- per-wave GEMM2 weight-row pointers (fc2 cols 0..511, expert 512..767) ----
    const WT* w2p[6];
#pragma unroll
    for (int nt = 0; nt < 6; ++nt) {
        int n = wave * 96 + nt * 16 + ln;          // 16-aligned tiles: branch uniform
        w2p[nt] = (n < SHRD)
            ? (w2 + (size_t)n * HDIM + half * 8)
            : (we + ((size_t)e * PDIM + (n - SHRD)) * HDIM + half * 8);
    }

    // output accumulators: 2 M-tiles x 6 N-tiles of 16x16 f32
    v8f acc[2][6];
#pragma unroll
    for (int m = 0; m < 2; ++m)
#pragma unroll
        for (int nt = 0; nt < 6; ++nt) acc[m][nt] = {};

    // this wave's fc1 row pointer (chunk 0), includes lane offsets
    const WT* w1base = w1 + (size_t)(wave * 16 + ln) * DDIM + half * 8;

    __syncthreads();

    for (int hc = 0; hc < HDIM / HC; ++hc) {
        // ---------- GEMM1: h[:, hc*HC + wave*16 .. +15] (reads xs + global only) ----------
        const WT* w1p = w1base + (size_t)hc * HC * DDIM;
        if (hc + 1 < HDIM / HC)
            __builtin_prefetch(w1base + (size_t)(hc + 1) * HC * DDIM, 0, 1);

        v8f a1[2]; a1[0] = {}; a1[1] = {};
        for (int k = 0; k < DDIM; k += 32) {
            v16bf bf = load_wfrag(w1p + k);
#pragma unroll
            for (int m = 0; m < 2; ++m) {
                v16bf af = load_wfrag(&xs[(m * 16 + ln) * XS + k + half * 8]);
                a1[m] = __builtin_amdgcn_wmma_f32_16x16x32_bf16(
                    false, af, false, bf, (short)0, a1[m], false, false);
            }
        }
        float bias1 = b1[hc * HC + wave * 16 + ln];

        __syncthreads();   // all waves done reading hs of previous chunk

        // ---------- bias + exact-erf GELU -> bf16 h-chunk in LDS ----------
        {
            int col = wave * 16 + ln;
#pragma unroll
            for (int m = 0; m < 2; ++m)
#pragma unroll
                for (int r = 0; r < 8; ++r) {
                    float v = gelu_erf(a1[m][r] + bias1);
                    hs[(m * 16 + half * 8 + r) * HS + col] = f2bf(v);
                }
        }
        __syncthreads();   // h-chunk visible to all waves

        // ---------- GEMM2: acc += h_chunk (32xHC) x W2chunk (HC x 96 per wave) ----------
        const size_t koff = (size_t)hc * HC;
        for (int k = 0; k < HC; k += 32) {
            v16bf af0 = load_wfrag(&hs[(ln) * HS + k + half * 8]);
            v16bf af1 = load_wfrag(&hs[(16 + ln) * HS + k + half * 8]);
#pragma unroll
            for (int nt = 0; nt < 6; ++nt) {
                v16bf bf = load_wfrag(w2p[nt] + koff + k);
                acc[0][nt] = __builtin_amdgcn_wmma_f32_16x16x32_bf16(
                    false, af0, false, bf, (short)0, acc[0][nt], false, false);
                acc[1][nt] = __builtin_amdgcn_wmma_f32_16x16x32_bf16(
                    false, af1, false, bf, (short)0, acc[1][nt], false, false);
            }
        }
    }

    // ---------- epilogue: add fc2/expert bias, store fp32 output ----------
    float* obase = out + (size_t)(b * SDIM + stile * MTILE) * DDIM;
#pragma unroll
    for (int m = 0; m < 2; ++m)
#pragma unroll
        for (int nt = 0; nt < 6; ++nt) {
            int n = wave * 96 + nt * 16 + ln;
            float bias = (n < SHRD) ? b2[n] : be[e * PDIM + (n - SHRD)];
#pragma unroll
            for (int r = 0; r < 8; ++r) {
                int row = m * 16 + half * 8 + r;
                obase[(size_t)row * DDIM + n] = acc[m][nt][r] + bias;
            }
        }
}

extern "C" void kernel_launch(void* const* d_in, const int* in_sizes, int n_in,
                              void* d_out, int out_size, void* d_ws, size_t ws_size,
                              hipStream_t stream) {
    (void)in_sizes; (void)n_in; (void)out_size;
    const float*     x   = (const float*)d_in[0];
    const long long* idx = (const long long*)d_in[1];
    const float*     w1  = (const float*)d_in[2];
    const float*     b1  = (const float*)d_in[3];
    const float*     w2  = (const float*)d_in[4];
    const float*     b2  = (const float*)d_in[5];
    const float*     we  = (const float*)d_in[6];
    const float*     be  = (const float*)d_in[7];
    float* out = (float*)d_out;

    constexpr size_t W1N = (size_t)HDIM * DDIM;          // 2,359,296
    constexpr size_t W2N = (size_t)SHRD * HDIM;          // 1,572,864
    constexpr size_t WEN = (size_t)6 * PDIM * HDIM;      // 4,718,592
    constexpr size_t TOT = W1N + W2N + WEN;              // 8,650,752 elems (17.3 MB bf16)

    dim3 grid(BDIM * (SDIM / MTILE));   // 384 workgroups
    dim3 block(256);                    // 8 wave32s

    if (ws_size >= TOT * sizeof(__bf16)) {
        // Pre-convert weights to bf16 once per call (deterministic), then run the
        // conversion-free WMMA kernel: hot loop is pure b128 loads + v_wmma.
        __bf16* wbf = (__bf16*)d_ws;
        cvt_f32_to_bf16<<<(int)((W1N / 8 + 255) / 256), 256, 0, stream>>>(w1, wbf, (int)(W1N / 8));
        cvt_f32_to_bf16<<<(int)((W2N / 8 + 255) / 256), 256, 0, stream>>>(w2, wbf + W1N, (int)(W2N / 8));
        cvt_f32_to_bf16<<<(int)((WEN / 8 + 255) / 256), 256, 0, stream>>>(we, wbf + W1N + W2N, (int)(WEN / 8));
        moe_mlp_fused<__bf16><<<grid, block, 0, stream>>>(
            x, idx, wbf, b1, wbf + W1N, b2, wbf + W1N + W2N, be, out);
    } else {
        // fallback: convert weights inline in the hot loop
        moe_mlp_fused<float><<<grid, block, 0, stream>>>(
            x, idx, w1, b1, w2, b2, we, be, out);
    }
}
